// SparseGPNoise_77197742178930
// MI455X (gfx1250) — compile-verified, matrix operand order checked by reading, not run on hardware
//
#include <hip/hip_runtime.h>
#include <hip/hip_bf16.h>
#include <math.h>

// ---------------- problem constants ----------------
#define DIMS    8
#define NSTEP   2048
#define MIND    64
#define TQ      1024
#define NCOL    1088          // 1024 query cols + 64 inducing cols
#define NPTS    2176          // t, t+dt, z, z+dt evaluation points
#define DT_C    1.0e-3f
#define DS_C    (1.0f/2048.0f)
#define SQRT_DT_C 0.0316227766016838f
#define SQRT_DS_OVER_DT 22.0970869120796f   // sqrt(1/2048)/1e-3
#define Z0_C    0.0055f
#define Z1_C    0.9945f

typedef __attribute__((ext_vector_type(16))) _Float16 v16h;
typedef __attribute__((ext_vector_type(8)))  _Float16 v8h;
typedef __attribute__((ext_vector_type(8)))  float    v8f;

__device__ __forceinline__ float sigmoidf_(float x) { return 1.0f / (1.0f + expf(-x)); }

// ---------------------------------------------------------------------------
// Kernel 1: Hurst MLP -> per (point, dim): 2h and 1/(sqrt(2h*ds)*Gamma(h+0.5))
// points: [0,1024)=t, [1024,2048)=t+dt, [2048,2112)=z, [2112,2176)=z+dt
// ---------------------------------------------------------------------------
__global__ void hurst_kernel(const float* __restrict__ t, const float* __restrict__ y0,
                             const float* __restrict__ W1, const float* __restrict__ b1,
                             const float* __restrict__ W2, const float* __restrict__ b2,
                             const float* __restrict__ Wc1, const float* __restrict__ bc1,
                             const float* __restrict__ Wc2, const float* __restrict__ bc2,
                             const float* __restrict__ dw, const float* __restrict__ cw,
                             float* __restrict__ twoH, float* __restrict__ invN) {
    int p = blockIdx.x * blockDim.x + threadIdx.x;
    if (p >= NPTS) return;
    float tv;
    if (p < TQ)            tv = t[p];
    else if (p < 2 * TQ)   tv = t[p - TQ] + DT_C;
    else {
        int m = (p - 2 * TQ) & 63;
        tv = Z0_C + (Z1_C - Z0_C) * (float)m * (1.0f / 63.0f);
        if (p >= 2 * TQ + MIND) tv += DT_C;
    }
    float y = y0[0];
    float hc[10], ht[10];
    float sn = sinf(tv), cs = cosf(tv);
#pragma unroll
    for (int j = 0; j < 10; ++j) {
        hc[j] = tanhf(y * Wc1[j] + bc1[j]);
        ht[j] = tanhf(W1[j * 3 + 0] * sn + W1[j * 3 + 1] * cs + W1[j * 3 + 2] * tv + b1[j]);
    }
    float dwv = dw[0], cwv = cw[0];
#pragma unroll
    for (int d = 0; d < DIMS; ++d) {
        float tc = b2[d], pc = bc2[d];
#pragma unroll
        for (int j = 0; j < 10; ++j) {
            tc += W2[d * 10 + j] * ht[j];
            pc += Wc2[d * 10 + j] * hc[j];
        }
        float h = sigmoidf_(sigmoidf_(tc) * dwv + sigmoidf_(pc) * cwv);
        twoH[p * DIMS + d] = 2.0f * h;
        invN[p * DIMS + d] = rsqrtf(2.0f * h * DS_C) / expf(lgammaf(h + 0.5f));
    }
}

// ---------------------------------------------------------------------------
// Kernel 2: increment weights -> Kall[d][s][n] (f16), n<1024 query, n>=1024 z
// ---------------------------------------------------------------------------
__global__ void weights_kernel(const float* __restrict__ t,
                               const float* __restrict__ twoH, const float* __restrict__ invN,
                               _Float16* __restrict__ Kall) {
    long idx = (long)blockIdx.x * blockDim.x + threadIdx.x;
    const long total = (long)DIMS * NSTEP * NCOL;
    if (idx >= total) return;
    int n = (int)(idx % NCOL);
    int s = (int)((idx / NCOL) % NSTEP);
    int d = (int)(idx / ((long)NCOL * NSTEP));
    float tv; int pA, pB;
    if (n < TQ) { tv = t[n]; pA = n; pB = n + TQ; }
    else {
        int m = n - TQ;
        tv = Z0_C + (Z1_C - Z0_C) * (float)m * (1.0f / 63.0f);
        pA = 2 * TQ + m; pB = 2 * TQ + MIND + m;
    }
    float s0 = (float)s * DS_C, s1 = s0 + DS_C;
    float th = twoH[pA * DIMS + d], in = invN[pA * DIMS + d];
    float d0 = tv - s0, d1 = tv - s1;
    float p0 = d0 > 0.f ? exp2f(th * log2f(d0)) : 0.f;
    float p1 = d1 > 0.f ? exp2f(th * log2f(d1)) : 0.f;
    float wa = sqrtf(p0 - p1 + 1e-12f) * in;
    float tb = tv + DT_C;
    th = twoH[pB * DIMS + d]; in = invN[pB * DIMS + d];
    d0 = tb - s0; d1 = tb - s1;
    p0 = d0 > 0.f ? exp2f(th * log2f(d0)) : 0.f;
    p1 = d1 > 0.f ? exp2f(th * log2f(d1)) : 0.f;
    float wb = sqrtf(p0 - p1 + 1e-12f) * in;
    wa = fmaxf(wa, 0.f); wb = fmaxf(wb, 0.f);
    float val = fmaxf((wb - wa) * SQRT_DS_OVER_DT, 0.f);
    Kall[((long)d * NSTEP + s) * NCOL + n] = (_Float16)val;
}

// ---------------------------------------------------------------------------
// Kernel 3a: big TN GEMM  C = A^T * B  (A:[K x M], B:[K x N], K-major, f16)
// 256 threads = 8 waves (2x4); block tile 128x128, BK=64.
// Staging: each thread owns one 8x8 (k x m) sub-block; 8 coalesced v8h global
// loads, in-register 8x8 half transpose, 8 x ds_store_b128 into m-major LDS.
// Each wave: 64x32 region = 4x2 WMMA tiles, 2 k-steps -> 16 WMMA per stage.
// mode 0: C = acc ; mode 1: C = sqrt(max(E - acc, 1e-12)) * scale
// Requires M,N multiples of 128, K multiple of 64.
// ---------------------------------------------------------------------------
#define LDK 72   // padded K pitch (halfs): 144B rows, 16B aligned sub-offsets
__global__ __launch_bounds__(256) void gemm128_tn_wmma(
    const _Float16* __restrict__ A, const _Float16* __restrict__ B,
    float* __restrict__ C, const float* __restrict__ E,
    int K, int ldA, int ldB, int ldC,
    long sA, long sB, long sC, long sE, int mode, float scale) {
    __shared__ __align__(16) _Float16 As[128][LDK];
    __shared__ __align__(16) _Float16 Bs[128][LDK];
    const int d = blockIdx.z;
    A += sA * d; B += sB * d; C += sC * d;
    if (E) E += sE * d;
    const int row0 = blockIdx.x * 128, col0 = blockIdx.y * 128;
    const int tid = threadIdx.x;
    const int lane = tid & 31, wv = tid >> 5;
    const int wr = wv >> 2, wc = wv & 3;        // 2x4 wave grid
    const int hi = lane >> 4, lm = lane & 15;   // WMMA lane decomposition
    // staging decomposition: threads [0,128) stage A, [128,256) stage B
    const int sg  = tid & 127;
    const int smb = sg & 15;          // m-block (8 cols of the tile)
    const int skb = sg >> 4;          // k-block (8 rows), 0..7
    const _Float16* gsrc = (tid < 128)
        ? (A + (size_t)(skb * 8) * ldA + row0 + smb * 8)
        : (B + (size_t)(skb * 8) * ldB + col0 + smb * 8);
    const size_t gld = (tid < 128) ? (size_t)ldA : (size_t)ldB;
    _Float16* lbase = (tid < 128) ? &As[smb * 8][skb * 8] : &Bs[smb * 8][skb * 8];

    v8f acc[4][2] = {};
    for (int k0 = 0; k0 < K; k0 += 64) {
        // ---- stage: 8x8 register transpose, vector LDS stores ----
        {
            v8h r[8];
#pragma unroll
            for (int j = 0; j < 8; ++j)
                r[j] = *(const v8h*)(gsrc + (size_t)(k0 + j) * gld);
#pragma unroll
            for (int i = 0; i < 8; ++i) {
                v8h w;
#pragma unroll
                for (int j = 0; j < 8; ++j) w[j] = r[j][i];
                *(v8h*)(lbase + (size_t)i * LDK) = w;
            }
        }
        __syncthreads();
#pragma unroll
        for (int ks = 0; ks < 2; ++ks) {
            // fragments: element e -> K = ks*32 + hi*8 + (e<8 ? e : e+8)
            v16h af[4], bf[2];
#pragma unroll
            for (int tm = 0; tm < 4; ++tm) {
                const _Float16* pa = &As[wr * 64 + tm * 16 + lm][ks * 32 + hi * 8];
                v8h lo = *(const v8h*)pa;
                v8h hh = *(const v8h*)(pa + 16);
#pragma unroll
                for (int e = 0; e < 8; ++e) { af[tm][e] = lo[e]; af[tm][e + 8] = hh[e]; }
            }
#pragma unroll
            for (int tn = 0; tn < 2; ++tn) {
                const _Float16* pb = &Bs[wc * 32 + tn * 16 + lm][ks * 32 + hi * 8];
                v8h lo = *(const v8h*)pb;
                v8h hh = *(const v8h*)(pb + 16);
#pragma unroll
                for (int e = 0; e < 8; ++e) { bf[tn][e] = lo[e]; bf[tn][e + 8] = hh[e]; }
            }
#pragma unroll
            for (int tm = 0; tm < 4; ++tm)
#pragma unroll
                for (int tn = 0; tn < 2; ++tn)
                    acc[tm][tn] = __builtin_amdgcn_wmma_f32_16x16x32_f16(
                        false, af[tm], false, bf[tn], (short)0, acc[tm][tn], false, false);
        }
        __syncthreads();
    }
    // epilogue: f32 C/D layout: m = r + 8*hi, n = lm
#pragma unroll
    for (int tm = 0; tm < 4; ++tm)
#pragma unroll
        for (int tn = 0; tn < 2; ++tn)
#pragma unroll
            for (int r = 0; r < 8; ++r) {
                int row = row0 + wr * 64 + tm * 16 + hi * 8 + r;
                int col = col0 + wc * 32 + tn * 16 + lm;
                float v = acc[tm][tn][r];
                size_t ci = (size_t)row * ldC + col;
                if (mode == 0) C[ci] = v;
                else C[ci] = sqrtf(fmaxf(E[(size_t)row * TQ + col] - v, 1e-12f)) * scale;
            }
}

// ---------------------------------------------------------------------------
// Kernel 3b: small TN GEMM (64x64 block tile, BK=32) for Ktz / Kzz shapes.
// 128 threads = 4 waves; each wave 32x32 = 2x2 WMMA tiles.
// Requires M,N multiples of 64, K multiple of 32.
// ---------------------------------------------------------------------------
#define LDSP 40
__global__ __launch_bounds__(128) void gemm64_tn_wmma(
    const _Float16* __restrict__ A, const _Float16* __restrict__ B,
    float* __restrict__ C,
    int K, int ldA, int ldB, int ldC,
    long sA, long sB, long sC) {
    __shared__ __align__(16) _Float16 As[64][LDSP];
    __shared__ __align__(16) _Float16 Bs[64][LDSP];
    const int d = blockIdx.z;
    A += sA * d; B += sB * d; C += sC * d;
    const int row0 = blockIdx.x * 64, col0 = blockIdx.y * 64;
    const int tid = threadIdx.x;
    const int lane = tid & 31, wv = tid >> 5;
    const int wr = wv >> 1, wc = wv & 1;
    const int hi = lane >> 4, lm = lane & 15;
    v8f acc[2][2] = {};
    for (int k0 = 0; k0 < K; k0 += 32) {
#pragma unroll
        for (int i = 0; i < 2; ++i) {
            int g = i * 128 + tid;              // 0..255
            int m0 = (g & 7) * 8;
            int kk = g >> 3;                    // 0..31
            v8h av = *(const v8h*)(A + (size_t)(k0 + kk) * ldA + row0 + m0);
            v8h bv = *(const v8h*)(B + (size_t)(k0 + kk) * ldB + col0 + m0);
#pragma unroll
            for (int e = 0; e < 8; ++e) { As[m0 + e][kk] = av[e]; Bs[m0 + e][kk] = bv[e]; }
        }
        __syncthreads();
        v16h af[2], bf[2];
#pragma unroll
        for (int tm = 0; tm < 2; ++tm) {
            const _Float16* pa = &As[wr * 32 + tm * 16 + lm][hi * 8];
            v8h lo = *(const v8h*)pa;
            v8h hh = *(const v8h*)(pa + 16);
#pragma unroll
            for (int e = 0; e < 8; ++e) { af[tm][e] = lo[e]; af[tm][e + 8] = hh[e]; }
        }
#pragma unroll
        for (int tn = 0; tn < 2; ++tn) {
            const _Float16* pb = &Bs[wc * 32 + tn * 16 + lm][hi * 8];
            v8h lo = *(const v8h*)pb;
            v8h hh = *(const v8h*)(pb + 16);
#pragma unroll
            for (int e = 0; e < 8; ++e) { bf[tn][e] = lo[e]; bf[tn][e + 8] = hh[e]; }
        }
#pragma unroll
        for (int tm = 0; tm < 2; ++tm)
#pragma unroll
            for (int tn = 0; tn < 2; ++tn)
                acc[tm][tn] = __builtin_amdgcn_wmma_f32_16x16x32_f16(
                    false, af[tm], false, bf[tn], (short)0, acc[tm][tn], false, false);
        __syncthreads();
    }
#pragma unroll
    for (int tm = 0; tm < 2; ++tm)
#pragma unroll
        for (int tn = 0; tn < 2; ++tn)
#pragma unroll
            for (int r = 0; r < 8; ++r) {
                int row = row0 + wr * 32 + tm * 16 + hi * 8 + r;
                int col = col0 + wc * 32 + tn * 16 + lm;
                C[(size_t)row * ldC + col] = acc[tm][tn][r];
            }
}

// ---------------------------------------------------------------------------
// Kernel 4: per-dim 64x64 Cholesky of Kzz + 1e-6 I  ->  L ; dBz = L * eps_z
// ---------------------------------------------------------------------------
__global__ __launch_bounds__(64) void chol_kernel(const float* __restrict__ Kzz,
                                                  float* __restrict__ L,
                                                  const float* __restrict__ eps_z,
                                                  float* __restrict__ dBz) {
    __shared__ float Am[64][65];
    int d = blockIdx.x, i = threadIdx.x;
    for (int j = 0; j < 64; ++j)
        Am[i][j] = Kzz[(size_t)d * 4096 + i * 64 + j] + (i == j ? 1e-6f : 0.f);
    __syncthreads();
    for (int k = 0; k < 64; ++k) {
        if (i == k) Am[k][k] = sqrtf(Am[k][k]);
        __syncthreads();
        if (i > k) Am[i][k] /= Am[k][k];
        __syncthreads();
        if (i > k)
            for (int j = k + 1; j <= i; ++j) Am[i][j] -= Am[i][k] * Am[j][k];
        __syncthreads();
    }
    float acc = 0.f;
    for (int j = 0; j < 64; ++j) {
        float lv = (j <= i) ? Am[i][j] : 0.f;
        L[(size_t)d * 4096 + i * 64 + j] = lv;
        acc += lv * eps_z[d * 64 + j];     // eps_z[d][j][0]
    }
    dBz[d * 64 + i] = acc;
}

// ---------------------------------------------------------------------------
// Kernel 5: per (d, t) cho_solve: L y = Ktz[t,:], L^T a = y
// writes alpha (f16, [d][m][t]) and mean -> d_out[d*1024 + t]
// ---------------------------------------------------------------------------
__global__ __launch_bounds__(256) void solve_kernel(const float* __restrict__ L,
                                                    const float* __restrict__ Ktz,
                                                    const float* __restrict__ dBz,
                                                    _Float16* __restrict__ alphaH,
                                                    float* __restrict__ outMean) {
    __shared__ float Ls[64][65];
    int d = blockIdx.y;
    int t = blockIdx.x * 256 + threadIdx.x;
    for (int idx = threadIdx.x; idx < 4096; idx += 256)
        Ls[idx >> 6][idx & 63] = L[(size_t)d * 4096 + idx];
    __syncthreads();
    float y[64];
    const float* r = Ktz + ((size_t)d * TQ + t) * 64;
#pragma unroll 1
    for (int i = 0; i < 64; ++i) {
        float s = r[i];
        for (int j = 0; j < i; ++j) s -= Ls[i][j] * y[j];
        y[i] = s / Ls[i][i];
    }
#pragma unroll 1
    for (int i = 63; i >= 0; --i) {
        float s = y[i];
        for (int j = i + 1; j < 64; ++j) s -= Ls[j][i] * y[j];
        y[i] = s / Ls[i][i];
    }
    float m = 0.f;
    for (int i = 0; i < 64; ++i) {
        alphaH[((size_t)d * 64 + i) * TQ + t] = (_Float16)y[i];
        m += y[i] * dBz[d * 64 + i];
    }
    outMean[d * TQ + t] = m;
}

// ---------------------------------------------------------------------------
extern "C" void kernel_launch(void* const* d_in, const int* in_sizes, int n_in,
                              void* d_out, int out_size, void* d_ws, size_t ws_size,
                              hipStream_t stream) {
    (void)in_sizes; (void)n_in; (void)out_size; (void)ws_size;
    const float* t    = (const float*)d_in[0];
    const float* y0   = (const float*)d_in[1];
    const float* epsz = (const float*)d_in[2];
    const float* W1   = (const float*)d_in[3];
    const float* b1   = (const float*)d_in[4];
    const float* W2   = (const float*)d_in[5];
    const float* b2   = (const float*)d_in[6];
    const float* Wc1  = (const float*)d_in[7];
    const float* bc1  = (const float*)d_in[8];
    const float* Wc2  = (const float*)d_in[9];
    const float* bc2  = (const float*)d_in[10];
    const float* dw   = (const float*)d_in[11];
    const float* cw   = (const float*)d_in[12];
    float* out = (float*)d_out;

    // workspace carve-out (256B aligned slices)
    size_t off = 0;
    auto take = [&](size_t bytes) -> void* {
        void* p = (char*)d_ws + off;
        off += (bytes + 255) & ~(size_t)255;
        return p;
    };
    float*     twoH  = (float*)    take((size_t)NPTS * DIMS * 4);
    float*     invN  = (float*)    take((size_t)NPTS * DIMS * 4);
    _Float16*  Kall  = (_Float16*) take((size_t)DIMS * NSTEP * NCOL * 2);   // 35.7 MB
    float*     Ktt   = (float*)    take((size_t)DIMS * TQ * TQ * 4);        // 33.6 MB
    float*     Ktz   = (float*)    take((size_t)DIMS * TQ * MIND * 4);      // 2 MB
    float*     Lz    = (float*)    take((size_t)DIMS * MIND * MIND * 4);
    float*     Kzz   = (float*)    take((size_t)DIMS * MIND * MIND * 4);
    float*     dBz   = (float*)    take((size_t)DIMS * MIND * 4);
    _Float16*  alphaH= (_Float16*) take((size_t)DIMS * MIND * TQ * 2);      // 1 MB

    // 1) Hurst coefficients
    hurst_kernel<<<(NPTS + 255) / 256, 256, 0, stream>>>(
        t, y0, W1, b1, W2, b2, Wc1, bc1, Wc2, bc2, dw, cw, twoH, invN);

    // 2) increment weights -> Kall (f16)
    {
        long total = (long)DIMS * NSTEP * NCOL;
        weights_kernel<<<(unsigned)((total + 255) / 256), 256, 0, stream>>>(t, twoH, invN, Kall);
    }

    const long sK = (long)NSTEP * NCOL;   // per-dim stride of Kall

    // 3) Ktt = Kt^T Kt   [1024 x 1024], K=2048  (big-tile WMMA kernel)
    gemm128_tn_wmma<<<dim3(TQ / 128, TQ / 128, DIMS), 256, 0, stream>>>(
        Kall, Kall, Ktt, nullptr, NSTEP, NCOL, NCOL, TQ,
        sK, sK, (long)TQ * TQ, 0, 0, 1.f);

    // 4) Ktz = Kt^T Kz   [1024 x 64], K=2048
    gemm64_tn_wmma<<<dim3(TQ / 64, 1, DIMS), 128, 0, stream>>>(
        Kall, Kall + TQ, Ktz, NSTEP, NCOL, NCOL, MIND,
        sK, sK, (long)TQ * MIND);

    // 5) Kzz = Kz^T Kz   [64 x 64], K=2048
    gemm64_tn_wmma<<<dim3(1, 1, DIMS), 128, 0, stream>>>(
        Kall + TQ, Kall + TQ, Kzz, NSTEP, NCOL, NCOL, MIND,
        sK, sK, (long)MIND * MIND);

    // 6) Cholesky + delta_Bz
    chol_kernel<<<DIMS, 64, 0, stream>>>(Kzz, Lz, epsz, dBz);

    // 7) triangular solves -> alpha (f16) + mean (first 8192 floats of d_out)
    solve_kernel<<<dim3(TQ / 256, DIMS), 256, 0, stream>>>(Lz, Ktz, dBz, alphaH, out);

    // 8) std = sqrt(max(Ktt - alpha^T alpha, 1e-12)) * sqrt(dt), fused epilogue
    gemm128_tn_wmma<<<dim3(TQ / 128, TQ / 128, DIMS), 256, 0, stream>>>(
        alphaH, alphaH, out + (size_t)DIMS * TQ, Ktt, MIND, TQ, TQ, TQ,
        (long)MIND * TQ, (long)MIND * TQ, (long)TQ * TQ, (long)TQ * TQ,
        1, SQRT_DT_C);
}
